// DeepRNNLayer_49366354100496
// MI455X (gfx1250) — matrix operand order repaired and checked by reading, not compiled, not run on hardware
//
#include <hip/hip_runtime.h>
#include <hip/hip_bf16.h>

#define B_   8
#define T_   2048
#define IN_  1024
#define H_   1024
#define G4_  4096           // 4*H
#define M_   (B_*T_)        // 16384

typedef __bf16 bf16;
typedef __attribute__((ext_vector_type(16))) __bf16 v16bf;
typedef __attribute__((ext_vector_type(8)))  __bf16 bf16x8;
typedef __attribute__((ext_vector_type(4)))  __bf16 bf16x4;
typedef __attribute__((ext_vector_type(8)))  float  v8f;

__device__ __forceinline__ v16bf join8(bf16x8 lo, bf16x8 hi) {
  v16bf r;
#pragma unroll
  for (int i = 0; i < 8; ++i) { r[i] = lo[i]; r[8 + i] = hi[i]; }
  return r;
}

__device__ __forceinline__ float sigmoidf_(float x) {
  return 1.0f / (1.0f + __expf(-x));
}
__device__ __forceinline__ float tanhf_(float x) {
  float e = __expf(2.0f * x);
  return (e - 1.0f) / (e + 1.0f);
}

// ---------------------------------------------------------------------------
// fp32 -> bf16 conversion (4-wide)
// ---------------------------------------------------------------------------
__global__ void cvt_bf16_kernel(const float* __restrict__ src,
                                bf16* __restrict__ dst, int n4) {
  int i = blockIdx.x * blockDim.x + threadIdx.x;
  if (i < n4) {
    float4 v = ((const float4*)src)[i];
    bf16x4 o;
    o[0] = (bf16)v.x; o[1] = (bf16)v.y; o[2] = (bf16)v.z; o[3] = (bf16)v.w;
    ((bf16x4*)dst)[i] = o;
  }
}

// ---------------------------------------------------------------------------
// init: h_state (16 x H bf16, rows 8..15 zero-padded for WMMA M=16),
//       c_state (8 x H f32), barrier counter
// ---------------------------------------------------------------------------
__global__ void lstm_init_kernel(const float* __restrict__ h0,
                                 const float* __restrict__ c0,
                                 bf16* __restrict__ h_state,
                                 float* __restrict__ c_state,
                                 int* __restrict__ ctr) {
  int tid = threadIdx.x;
  for (int i = tid; i < 16 * H_; i += blockDim.x) {
    int b = i / H_;
    h_state[i] = (b < B_) ? (bf16)h0[i] : (bf16)0.0f;
  }
  for (int i = tid; i < B_ * H_; i += blockDim.x) c_state[i] = c0[i];
  if (tid == 0) *ctr = 0;
}

// ---------------------------------------------------------------------------
// Big GEMM: gates_x[M=16384][4096] = x_bf16[M][1024] * W_ih[4096][1024]^T
//           + (b_ih + b_hh)
// Block: 256 threads (8 waves). Block tile 128(M) x 128(N), K-chunks of 32.
// Waves arranged 2(M) x 4(N); each wave owns a 64x32 tile = 4x2 WMMA tiles:
// per K-chunk 12 ds_load_b128 feed 8 v_wmma (vs 8:4 before).
// ---------------------------------------------------------------------------
__launch_bounds__(256)
__global__ void gemm_gates_kernel(const bf16* __restrict__ X,
                                  const bf16* __restrict__ W,
                                  const float* __restrict__ b_ih,
                                  const float* __restrict__ b_hh,
                                  float* __restrict__ G) {
  __shared__ bf16 As[128][48];   // 48-elem stride: 96B rows, 16B aligned cols
  __shared__ bf16 Bs[128][48];

  const int tid  = threadIdx.x;
  const int lane = tid & 31;
  const int wave = tid >> 5;
  const int wm   = wave & 1;     // M half (64 rows each)
  const int wn   = wave >> 1;    // N quadrant (32 cols each)
  const int lm   = lane & 15;
  const bool hi  = lane >= 16;
  const int m0   = blockIdx.y * 128;
  const int n0   = blockIdx.x * 128;

  v8f acc[4][2] = {};

  for (int kc = 0; kc < IN_ / 32; ++kc) {
    const int k0 = kc * 32;
    // cooperative loads: A and B tiles are each 128x32 bf16 -> 2 b128/thread
    {
      int g = tid;
#pragma unroll
      for (int rep = 0; rep < 2; ++rep) {
        int row = g >> 2;
        int col = (g & 3) * 8;
        *(bf16x8*)&As[row][col] =
            *(const bf16x8*)&X[(size_t)(m0 + row) * IN_ + k0 + col];
        *(bf16x8*)&Bs[row][col] =
            *(const bf16x8*)&W[(size_t)(n0 + row) * IN_ + k0 + col];
        g += 256;
      }
    }
    __syncthreads();

    v16bf a[4], b[2];
#pragma unroll
    for (int i = 0; i < 4; ++i) {
      // A 16x32 bf16 layout: lanes 0-15 hold K 0..7 & 16..23, lanes 16-31
      // hold K 8..15 & 24..31 (row M = lane%16)
      const bf16* ap = &As[wm * 64 + i * 16 + lm][hi ? 8 : 0];
      a[i] = join8(*(const bf16x8*)ap, *(const bf16x8*)(ap + 16));
    }
#pragma unroll
    for (int j = 0; j < 2; ++j) {
      // B 32x16 bf16 layout: lanes 0-15 hold K 0..7 & 16..23, lanes 16-31
      // hold K 8..15 & 24..31 (col N = lane%16)
      const bf16* bp = &Bs[wn * 32 + j * 16 + lm][hi ? 8 : 0];
      b[j] = join8(*(const bf16x8*)bp, *(const bf16x8*)(bp + 16));
    }
#pragma unroll
    for (int i = 0; i < 4; ++i)
#pragma unroll
      for (int j = 0; j < 2; ++j)
        acc[i][j] = __builtin_amdgcn_wmma_f32_16x16x32_bf16(
            false, a[i], false, b[j], (short)0, acc[i][j], false, false);
    __syncthreads();
  }

  // epilogue: + (b_ih + b_hh), C layout: vgpr r, lanes 0-15 -> M=r,
  // lanes 16-31 -> M=r+8, N = lane%16
#pragma unroll
  for (int j = 0; j < 2; ++j) {
    const int gn = n0 + wn * 32 + j * 16 + lm;
    const float bias = b_ih[gn] + b_hh[gn];
#pragma unroll
    for (int i = 0; i < 4; ++i) {
#pragma unroll
      for (int r = 0; r < 8; ++r) {
        const int gm = m0 + wm * 64 + i * 16 + r + (hi ? 8 : 0);
        G[(size_t)gm * G4_ + gn] = acc[i][j][r] + bias;
      }
    }
  }
}

// ---------------------------------------------------------------------------
// Persistent recurrent kernel. 32 blocks x 256 threads = 256 waves, one wave
// per 16-wide gate tile (4 gates x 64 h-tiles). W_hh is time-invariant and a
// wave's B-slice is 32 fragments = 256 VGPRs of bf16 -- CDNA5 allows 1024
// VGPRs/wave, so the whole slice is preloaded into registers ONCE and the
// per-step inner loop is just h-state loads + 32 back-to-back v_wmma.
// The four gates of a given 16-unit slice live in the same block -> fused
// cell update through LDS. Grid-wide release/acquire spin barrier per step.
// ---------------------------------------------------------------------------
__launch_bounds__(256)
__global__ void lstm_recurrent_kernel(const float* __restrict__ Gx,
                                      const bf16* __restrict__ Whh,
                                      bf16* __restrict__ h_state,
                                      float* __restrict__ c_state,
                                      float* __restrict__ out,
                                      float* __restrict__ hT,
                                      float* __restrict__ cT,
                                      int* __restrict__ ctr) {
  __shared__ float tiles[8][16][17];   // per-wave 16x16 f32 gate tiles

  const int tid   = threadIdx.x;
  const int lane  = tid & 31;
  const int wave  = tid >> 5;
  const int gate  = wave & 3;                 // i,f,g,o
  const int htl   = wave >> 2;                // 0/1 within block
  const int htile = blockIdx.x * 2 + htl;     // 0..63
  const int lm    = lane & 15;
  const bool hi   = lane >= 16;
  const int n0    = gate * H_ + htile * 16;
  const int ncol  = n0 + lm;

  // cell-update mapping: 256 threads <-> 2 h-tiles x 8 batch x 16 units
  const int cu_u = tid & 15;
  const int cu_b = (tid >> 4) & 7;
  const int cu_t = tid >> 7;
  const int cu_ug = (blockIdx.x * 2 + cu_t) * 16 + cu_u;
  const int cidx  = cu_b * H_ + cu_ug;

  const int nblocks = gridDim.x;

  // --- preload this wave's W_hh slice into registers (persistent weights) ---
  v16bf bfrag[H_ / 32];                       // 32 x 8 VGPRs = 256 VGPRs
  {
    const bf16* brow = Whh + (size_t)ncol * H_;
#pragma unroll
    for (int kc = 0; kc < H_ / 32; ++kc) {
      const bf16* bp = brow + kc * 32 + (hi ? 8 : 0);
      bfrag[kc] = join8(*(const bf16x8*)bp, *(const bf16x8*)(bp + 16));
    }
  }

  const bf16* arow = h_state + (size_t)lm * H_;

  for (int t = 0; t < T_; ++t) {
    // seed accumulator with gates_x (biases already folded in)
    v8f acc;
#pragma unroll
    for (int r = 0; r < 8; ++r) {
      const int bm = r + (hi ? 8 : 0);        // batch row (rows 8..15 pad)
      acc[r] = (bm < B_) ? Gx[((size_t)(bm * T_ + t)) * G4_ + ncol] : 0.0f;
    }

    // acc += h[16x1024] * Whh_slice^T : only h-state traffic in the loop
#pragma unroll
    for (int kc = 0; kc < H_ / 32; ++kc) {
      const bf16* ap = arow + kc * 32 + (hi ? 8 : 0);
      v16bf a = join8(*(const bf16x8*)ap, *(const bf16x8*)(ap + 16));
      acc = __builtin_amdgcn_wmma_f32_16x16x32_bf16(
          false, a, false, bfrag[kc], (short)0, acc, false, false);
    }

    // publish this wave's gate tile
#pragma unroll
    for (int r = 0; r < 8; ++r)
      tiles[wave][r + (hi ? 8 : 0)][lm] = acc[r];
    __syncthreads();

    // fused LSTM cell update (one element per thread)
    {
      const float gi = tiles[cu_t * 4 + 0][cu_b][cu_u];
      const float gf = tiles[cu_t * 4 + 1][cu_b][cu_u];
      const float gg = tiles[cu_t * 4 + 2][cu_b][cu_u];
      const float go = tiles[cu_t * 4 + 3][cu_b][cu_u];
      const float i_ = sigmoidf_(gi);
      const float f_ = sigmoidf_(gf);
      const float g_ = tanhf_(gg);
      const float o_ = sigmoidf_(go);
      const float c_ = f_ * c_state[cidx] + i_ * g_;
      const float h_ = o_ * tanhf_(c_);
      c_state[cidx] = c_;
      h_state[cidx] = (bf16)h_;
      out[((size_t)(cu_b * T_ + t)) * H_ + cu_ug] = h_;
      if (t == T_ - 1) { hT[cidx] = h_; cT[cidx] = c_; }
    }

    // device-wide barrier (release h/c state, acquire for next step)
    __threadfence();
    __syncthreads();
    if (tid == 0) {
      __hip_atomic_fetch_add(ctr, 1, __ATOMIC_RELEASE,
                             __HIP_MEMORY_SCOPE_AGENT);
      const int target = (t + 1) * nblocks;
      while (__hip_atomic_load(ctr, __ATOMIC_ACQUIRE,
                               __HIP_MEMORY_SCOPE_AGENT) < target)
        __builtin_amdgcn_s_sleep(2);
    }
    __syncthreads();
    __threadfence();
  }
}

// ---------------------------------------------------------------------------
extern "C" void kernel_launch(void* const* d_in, const int* in_sizes, int n_in,
                              void* d_out, int out_size, void* d_ws,
                              size_t ws_size, hipStream_t stream) {
  (void)in_sizes; (void)n_in; (void)out_size; (void)ws_size;
  const float* x   = (const float*)d_in[0];   // [B,T,IN]
  const float* h0  = (const float*)d_in[1];   // [B,H]
  const float* c0  = (const float*)d_in[2];   // [B,H]
  const float* Wih = (const float*)d_in[3];   // [4H,IN]
  const float* Whh = (const float*)d_in[4];   // [4H,H]
  const float* bih = (const float*)d_in[5];   // [4H]
  const float* bhh = (const float*)d_in[6];   // [4H]
  float* out = (float*)d_out;                 // [B,T,H] ++ h_T ++ c_T

  char* ws = (char*)d_ws;
  size_t off = 0;
  bf16*  xb   = (bf16*)(ws + off);  off += (size_t)M_ * IN_ * 2;   // 32 MB
  bf16*  Wihb = (bf16*)(ws + off);  off += (size_t)G4_ * IN_ * 2;  //  8 MB
  bf16*  Whhb = (bf16*)(ws + off);  off += (size_t)G4_ * H_ * 2;   //  8 MB
  float* Gx   = (float*)(ws + off); off += (size_t)M_ * G4_ * 4;   // 256 MB
  bf16*  hs   = (bf16*)(ws + off);  off += (size_t)16 * H_ * 2;    // 32 KB
  float* cs   = (float*)(ws + off); off += (size_t)B_ * H_ * 4;    // 32 KB
  int*   ctr  = (int*)(ws + off);   off += 256;

  float* hT = out + (size_t)B_ * T_ * H_;
  float* cT = hT + (size_t)B_ * H_;

  lstm_init_kernel<<<dim3(1), dim3(256), 0, stream>>>(h0, c0, hs, cs, ctr);

  int n4 = M_ * IN_ / 4;
  cvt_bf16_kernel<<<dim3((n4 + 255) / 256), dim3(256), 0, stream>>>(x, xb, n4);
  n4 = G4_ * IN_ / 4;
  cvt_bf16_kernel<<<dim3((n4 + 255) / 256), dim3(256), 0, stream>>>(Wih, Wihb, n4);
  n4 = G4_ * H_ / 4;
  cvt_bf16_kernel<<<dim3((n4 + 255) / 256), dim3(256), 0, stream>>>(Whh, Whhb, n4);

  gemm_gates_kernel<<<dim3(G4_ / 128, M_ / 128), dim3(256), 0, stream>>>(
      xb, Wihb, bih, bhh, Gx);

  lstm_recurrent_kernel<<<dim3(32), dim3(256), 0, stream>>>(
      Gx, Whhb, hs, cs, out, hT, cT, ctr);
}
